// LSTM_single_task_76647986364859
// MI455X (gfx1250) — matrix operand order repaired and checked by reading, not compiled, not run on hardware
//
#include <hip/hip_runtime.h>
#include <math.h>

typedef __attribute__((ext_vector_type(16))) __bf16 v16bf;
typedef __attribute__((ext_vector_type(8)))  float  v8f;

#define BSZ 32768
#define HD  512
#define ID  512
#define KTOT 1024    // concatenated K: [x | h]
#define MT_BLK 128   // block M tile
#define NT_BLK 64    // block N tile
#define KT 32        // K step (bf16 WMMA depth)
#define NKB (KTOT / KT)   // 32 K-steps

#ifndef USE_ASYNC_LDS
#define USE_ASYNC_LDS 1
#endif

static __device__ __forceinline__ unsigned short f2bf(float f) {
    unsigned int u = __builtin_bit_cast(unsigned int, f);
    unsigned int r = (u + 0x7FFFu + ((u >> 16) & 1u)) >> 16;  // RNE truncate
    return (unsigned short)r;
}

static __device__ __forceinline__ unsigned int pack2(float a, float b) {
    return (unsigned int)f2bf(a) | ((unsigned int)f2bf(b) << 16);
}

static __device__ __forceinline__ float fast_tanh(float x) {
#if __has_builtin(__builtin_amdgcn_tanhf)
    return __builtin_amdgcn_tanhf(x);
#elif __has_builtin(__builtin_amdgcn_tanh_f32)
    return __builtin_amdgcn_tanh_f32(x);
#else
    return tanhf(x);
#endif
}

static __device__ __forceinline__ float sigmoidf_fast(float x) {
    return 1.0f / (1.0f + __expf(-x));
}

// 16-byte global -> LDS copy; async path on gfx1250 (tracked with ASYNCcnt)
static __device__ __forceinline__ void copy16_g2l(unsigned short* lds_dst,
                                                  const unsigned short* gsrc) {
#if USE_ASYNC_LDS
    unsigned int       ldso = (unsigned int)(uintptr_t)lds_dst;       // LDS byte offset
    unsigned long long ga   = (unsigned long long)(uintptr_t)gsrc;
    asm volatile("global_load_async_to_lds_b128 %0, %1, off"
                 :: "v"(ldso), "v"(ga) : "memory");
#else
    *(uint4*)lds_dst = *(const uint4*)gsrc;
#endif
}

static __device__ __forceinline__ void wait_async_copies() {
#if USE_ASYNC_LDS
#if __has_builtin(__builtin_amdgcn_s_wait_asynccnt)
    __builtin_amdgcn_s_wait_asynccnt(0);
#else
    asm volatile("s_wait_asynccnt 0" ::: "memory");
#endif
#endif
}

// ---------- prep kernel 1: f32 [32768 x 512] -> bf16 into Acat[32768][1024] at col offset ----------
__global__ __launch_bounds__(256) void cvt_rows_bf16(
    const float* __restrict__ src, unsigned short* __restrict__ dst, int dstColOff)
{
    size_t tid = (size_t)blockIdx.x * blockDim.x + threadIdx.x;  // 8 elems each
    int r  = (int)(tid >> 6);        // 64 chunks of 8 per 512-wide row
    int c0 = (int)(tid & 63) * 8;
    const float* s = src + (size_t)r * ID + c0;
    float4 f0 = ((const float4*)s)[0];
    float4 f1 = ((const float4*)s)[1];
    uint4 o;
    o.x = pack2(f0.x, f0.y); o.y = pack2(f0.z, f0.w);
    o.z = pack2(f1.x, f1.y); o.w = pack2(f1.z, f1.w);
    *(uint4*)(dst + (size_t)r * KTOT + dstColOff + c0) = o;
}

// ---------- prep kernel 2: build WT[g][n][k] bf16, n-major, k = [Wx | Wh] ----------
__global__ __launch_bounds__(256) void build_wt(
    const float* __restrict__ Wx0, const float* __restrict__ Wh0,
    const float* __restrict__ Wx1, const float* __restrict__ Wh1,
    const float* __restrict__ Wx2, const float* __restrict__ Wh2,
    unsigned short* __restrict__ WT)
{
    const int g = blockIdx.y;
    const float* Wx = (g == 0) ? Wx0 : ((g == 1) ? Wx1 : Wx2);
    const float* Wh = (g == 0) ? Wh0 : ((g == 1) ? Wh1 : Wh2);
    int tid = blockIdx.x * 256 + threadIdx.x;   // 512*128 = 65536 total
    int n  = tid >> 7;                          // 0..511
    int k0 = (tid & 127) * 8;                   // 0..1016, chunk within one matrix
    const float* Wsrc = (k0 < 512) ? Wx : Wh;
    int kk = (k0 < 512) ? k0 : (k0 - 512);
    float v[8];
#pragma unroll
    for (int j = 0; j < 8; ++j) v[j] = Wsrc[(size_t)(kk + j) * HD + n];
    uint4 o;
    o.x = pack2(v[0], v[1]); o.y = pack2(v[2], v[3]);
    o.z = pack2(v[4], v[5]); o.w = pack2(v[6], v[7]);
    *(uint4*)(WT + (size_t)g * HD * KTOT + (size_t)n * KTOT + k0) = o;
}

// ---------- main fused GEMM (3 gates) + LSTM activation epilogue ----------
__global__ __launch_bounds__(256) void lstm_wmma_bf16(
    const unsigned short* __restrict__ Acat,   // [32768][1024] bf16
    const unsigned short* __restrict__ WT,     // [3][512][1024] bf16 n-major
    const float* __restrict__ bxi, const float* __restrict__ bxc,
    const float* __restrict__ bxo,
    float* __restrict__ out_h, float* __restrict__ out_c)
{
    // double-buffered tiles: A row-major [row][k], B col-major [n][k]
    __shared__ __align__(16) unsigned short As[2][MT_BLK * KT];       // 2 x 8 KB
    __shared__ __align__(16) unsigned short Bs[2][3][NT_BLK * KT];    // 2 x 12 KB

    const int t     = threadIdx.x;
    const int lane  = t & 31;
    const int wave  = t >> 5;      // 0..7
    const int wm    = wave >> 1;   // 0..3 (M)
    const int wn    = wave & 1;    // 0..1 (N)
    const int l15   = lane & 15;
    const int khalf = lane >> 4;   // 0/1

    const int rowBase = blockIdx.x * MT_BLK;
    const int colBase = blockIdx.y * NT_BLK;

    v8f acc[3][2][2];
    const v8f vzero = {0.f, 0.f, 0.f, 0.f, 0.f, 0.f, 0.f, 0.f};
#pragma unroll
    for (int g = 0; g < 3; ++g)
#pragma unroll
        for (int a = 0; a < 2; ++a)
#pragma unroll
            for (int b = 0; b < 2; ++b) acc[g][a][b] = vzero;

    // chunk coordinates (16B granules, 8 bf16 each)
    const int cr = t >> 2;        // 0..63 (row / n index)
    const int cq = (t & 3) * 8;   // 0/8/16/24 (k offset within tile)

    // per-thread staging base pointers (loop adds kBase only)
    const unsigned short* aSrc0 = Acat + (size_t)(rowBase + cr) * KTOT + cq;
    const unsigned short* aSrc1 = Acat + (size_t)(rowBase + cr + 64) * KTOT + cq;
    const unsigned short* bSrc[3];
#pragma unroll
    for (int g = 0; g < 3; ++g)
        bSrc[g] = WT + (size_t)g * HD * KTOT + (size_t)(colBase + cr) * KTOT + cq;

#define STAGE_TILES(buf, kBase)                                            \
    do {                                                                   \
        copy16_g2l(&As[(buf)][cr * KT + cq],        aSrc0 + (kBase));      \
        copy16_g2l(&As[(buf)][(cr + 64) * KT + cq], aSrc1 + (kBase));      \
        _Pragma("unroll")                                                  \
        for (int g = 0; g < 3; ++g)                                        \
            copy16_g2l(&Bs[(buf)][g][cr * KT + cq], bSrc[g] + (kBase));    \
    } while (0)

    // prologue: stage tile 0
    STAGE_TILES(0, 0);
    wait_async_copies();
    __syncthreads();

    for (int kb = 0; kb < NKB; ++kb) {
        const int cur = kb & 1;

        // ---- overlap: async-stage next tile into the other buffer ----
        if (kb + 1 < NKB) STAGE_TILES(cur ^ 1, (kb + 1) * KT);
        if (kb + 2 < NKB) __builtin_prefetch(aSrc0 + (kb + 2) * KT, 0, 1);

        // ---- load ALL fragments up front (batched dscnt wait) ----
        // A: ISA 16-bit A 16x32 layout -> two 16B chunks per fragment
        v16bf afr[2];
#pragma unroll
        for (int mt = 0; mt < 2; ++mt) {
            const unsigned short* p =
                &As[cur][(wm * 32 + mt * 16 + l15) * KT + khalf * 8];
            union { uint4 u[2]; v16bf v; } fu;
            fu.u[0] = *(const uint4*)p;         // k = khalf*8 + 0..7
            fu.u[1] = *(const uint4*)(p + 16);  // k = 16 + khalf*8 + 0..7
            afr[mt] = fu.v;
        }
        // B: lane -> col l&15, k = khalf*8 + 0..7 and 16 + khalf*8 + 0..7
        v16bf bfr[3][2];
#pragma unroll
        for (int g = 0; g < 3; ++g)
#pragma unroll
            for (int nt = 0; nt < 2; ++nt) {
                const unsigned short* p =
                    &Bs[cur][g][(wn * 32 + nt * 16 + l15) * KT + khalf * 8];
                union { uint4 u[2]; v16bf v; } fu;
                fu.u[0] = *(const uint4*)p;
                fu.u[1] = *(const uint4*)(p + 16);
                bfr[g][nt] = fu.v;
            }

        // ---- 12 back-to-back WMMAs ----
#pragma unroll
        for (int g = 0; g < 3; ++g)
#pragma unroll
            for (int nt = 0; nt < 2; ++nt)
#pragma unroll
                for (int mt = 0; mt < 2; ++mt) {
                    acc[g][mt][nt] = __builtin_amdgcn_wmma_f32_16x16x32_bf16(
                        false, afr[mt], false, bfr[g][nt],
                        (short)0, acc[g][mt][nt], false, false);
                }

        // next tile's async copies must be in LDS; all waves done reading cur
        wait_async_copies();
        __syncthreads();
    }
#undef STAGE_TILES

    // ---- epilogue: biases + activations -> h_next, c_next ----
#pragma unroll
    for (int nt = 0; nt < 2; ++nt) {
        const int col = colBase + wn * 32 + nt * 16 + l15;
        const float bi = bxi[col];
        const float bc = bxc[col];
        const float bo = bxo[col];
#pragma unroll
        for (int mt = 0; mt < 2; ++mt) {
            const int row0 = rowBase + wm * 32 + mt * 16 + khalf * 8;
#pragma unroll
            for (int v = 0; v < 8; ++v) {
                const float pi = acc[0][mt][nt][v] + bi;
                const float pc = acc[1][mt][nt][v] + bc;
                const float po = acc[2][mt][nt][v] + bo;
                const float ig = sigmoidf_fast(pi);
                const float og = sigmoidf_fast(po);
                const float cn = ig * fast_tanh(pc);
                const float hn = og * fast_tanh(cn);
                const size_t idx = (size_t)(row0 + v) * HD + col;
                out_h[idx] = hn;
                out_c[idx] = cn;
            }
        }
    }
}

extern "C" void kernel_launch(void* const* d_in, const int* in_sizes, int n_in,
                              void* d_out, int out_size, void* d_ws, size_t ws_size,
                              hipStream_t stream) {
    // setup_inputs order:
    // 0 x, 1 h, 2 c_prev, 3 Wxi, 4 bxi, 5 Whi, 6 Wxf, 7 bxf, 8 Whf,
    // 9 Wxc, 10 bxc, 11 Whc, 12 Wxo, 13 bxo, 14 Who
    const float* x   = (const float*)d_in[0];
    const float* h   = (const float*)d_in[1];
    const float* Wxi = (const float*)d_in[3];
    const float* bxi = (const float*)d_in[4];
    const float* Whi = (const float*)d_in[5];
    const float* Wxc = (const float*)d_in[9];
    const float* bxc = (const float*)d_in[10];
    const float* Whc = (const float*)d_in[11];
    const float* Wxo = (const float*)d_in[12];
    const float* bxo = (const float*)d_in[13];
    const float* Who = (const float*)d_in[14];

    // scratch layout: Acat bf16 [32768][1024], then WT bf16 [3][512][1024] (~70 MB)
    unsigned short* Acat = (unsigned short*)d_ws;
    unsigned short* WT   = Acat + (size_t)BSZ * KTOT;

    float* out_h = (float*)d_out;
    float* out_c = out_h + (size_t)BSZ * HD;

    // prep: convert/concat activations, convert/transpose weights
    {
        int blocks = (BSZ * (ID / 8)) / 256;   // 8192
        cvt_rows_bf16<<<blocks, 256, 0, stream>>>(x, Acat, 0);
        cvt_rows_bf16<<<blocks, 256, 0, stream>>>(h, Acat, ID);
        dim3 wgrid((HD * (KTOT / 8)) / 256, 3);  // (256, 3)
        build_wt<<<wgrid, 256, 0, stream>>>(Wxi, Whi, Wxc, Whc, Wxo, Who, WT);
    }

    // main fused GEMM + activations
    dim3 grid(BSZ / MT_BLK, HD / NT_BLK);   // 256 x 8
    lstm_wmma_bf16<<<grid, 256, 0, stream>>>(
        Acat, WT, bxi, bxc, bxo, out_h, out_c);
}
